// MessageControlGraphAttentionLayer_2267742732711
// MI455X (gfx1250) — matrix operand order repaired and checked by reading, not compiled, not run on hardware
//
#include <hip/hip_runtime.h>
#include <hip/hip_bf16.h>
#include <math.h>

// ---------------- problem constants ----------------
#define Bc 4
#define Tc 256
#define Dc 256
#define Oc 256
#define Hc 8
#define BT (Bc*Tc)          // 1024 rows
#define BN_EPS 1e-5f
#define SELU_SCALE 1.0507009873554805f
#define SELU_ALPHA 1.6732632423543772f

// ---------------- WMMA vector types ----------------
typedef __bf16  v16bf __attribute__((ext_vector_type(16)));
typedef __bf16  v8bf  __attribute__((ext_vector_type(8)));
typedef float   v8f   __attribute__((ext_vector_type(8)));

// LDS pair/a buffer: 64 rows x 256 cols bf16, padded stride (keeps 16B align, breaks bank stride)
#define PSTR 264

// ---------------- workspace layout (bytes) ----------------
#define WS_SPAD   0                     // int[4*257]
#define WS_WSWZ   8192                  // bf16[8*16*32*16] = 65536 elems
#define WS_AWSWZ  139264                // bf16[8*32*16]    = 4096 elems
#define WS_Y      147456                // float[1024*256]
#define WS_SUM    1196032               // float[256]
#define WS_SUMSQ  1197056               // float[256]

__device__ __forceinline__ __bf16 f2bf(float f) {
    unsigned int u = __float_as_uint(f);
    unsigned int r = (u + 0x7FFFu + ((u >> 16) & 1u)) >> 16;
    union { unsigned short s; __bf16 b; } cv; cv.s = (unsigned short)r;
    return cv.b;
}

__device__ __forceinline__ v16bf join8(v8bf lo, v8bf hi) {
    union { v16bf v; v8bf h[2]; } u;
    u.h[0] = lo; u.h[1] = hi;
    return u.v;
}

// ============ prep: per-batch cumsum of boundary -> spad[b][0..T] ============
__global__ void k_prep_spad(const int* __restrict__ boundary, int* __restrict__ spad) {
    int b = blockIdx.x;
    if (threadIdx.x == 0) {
        int s = 0;
        spad[b * (Tc + 1)] = 0;
        for (int t = 0; t < Tc; ++t) {
            s += boundary[b * Tc + t];
            spad[b * (Tc + 1) + t + 1] = s;
        }
    }
}

// ============ prep: swizzle weights into per-lane WMMA-B fragment layout ======
// W_swz[ks][nt][lane][e]  : value = att_proj_w[(ks*32 + (e>>3)*16 + (lane>>4)*8 + (e&7))*Oc + nt*16 + (lane&15)]
// aw_swz[ks][lane][e]     : value = (n<8) ? att_weight[(ks*32 + (e>>3)*16 + (lane>>4)*8 + (e&7))*Hc + n] : 0
__global__ void k_prep_w(const float* __restrict__ wproj, const float* __restrict__ aw,
                         __bf16* __restrict__ wswz, __bf16* __restrict__ awswz) {
    int idx = blockIdx.x * blockDim.x + threadIdx.x;
    if (idx < 65536) {
        int e    = idx & 15;
        int lane = (idx >> 4) & 31;
        int nt   = (idx >> 9) & 15;
        int ks   = idx >> 13;
        int kk   = ks * 32 + ((e >> 3) << 4) + ((lane >> 4) << 3) + (e & 7);
        int oo   = nt * 16 + (lane & 15);
        wswz[idx] = f2bf(wproj[kk * Oc + oo]);
    } else if (idx < 65536 + 4096) {
        int j2   = idx - 65536;
        int e    = j2 & 15;
        int lane = (j2 >> 4) & 31;
        int ks   = j2 >> 9;
        int kk   = ks * 32 + ((e >> 3) << 4) + ((lane >> 4) << 3) + (e & 7);
        int n    = lane & 15;
        awswz[j2] = (n < Hc) ? f2bf(aw[kk * Hc + n]) : f2bf(0.0f);
    }
}

// ============ main fused attention kernel: one block per (b, j) ==============
__global__ void __launch_bounds__(256)
k_attn(const float* __restrict__ x,
       const float* __restrict__ att_proj_b,
       const float* __restrict__ proj_att_w, const float* __restrict__ proj_att_b,
       const float* __restrict__ proj_no_w,  const float* __restrict__ proj_no_b,
       const __bf16* __restrict__ wswz, const __bf16* __restrict__ awswz,
       const int* __restrict__ spad,
       float* __restrict__ yout) {

    __shared__ float xj[Dc];
    __shared__ alignas(16) __bf16 pbuf[64 * PSTR];  // pair matrix, then reused for tanh(a) bf16
    __shared__ float attL[Tc * Hc];                 // logits -> softmax weights (in place)
    __shared__ float x1s[Dc * Hc];                  // aggregated x1 row (flat d*H+h)

    const int tid  = threadIdx.x;
    const int lane = tid & 31;
    const int wave = tid >> 5;
    const int blk  = blockIdx.x;
    const int b    = blk / Tc;
    const int j    = blk % Tc;

    const int half   = lane >> 4;           // 0/1
    const int lrow   = lane & 15;
    const int mt     = wave & 3;            // M-tile within 64-row super-tile
    const int nhalf  = wave >> 2;           // which 128-wide N half
    const int nbase  = nhalf * 128;
    const int* sp    = spad + b * (Tc + 1);

    xj[tid] = x[(b * Tc + j) * Dc + tid];
    __syncthreads();

    // ---- loop over 4 super-tiles of 64 keys ----
    #pragma unroll 1
    for (int st = 0; st < 4; ++st) {
        const int k0 = st * 64;

        // build pair matrix: pbuf[m][d] = bf16(xj[d] * x[b][k0+m][d])
        #pragma unroll 1
        for (int m = 0; m < 64; ++m) {
            float xv = x[(b * Tc + k0 + m) * Dc + tid];
            pbuf[m * PSTR + tid] = f2bf(xj[tid] * xv);
        }
        __syncthreads();

        // ---- GEMM1: (64 x 256) @ (256 x 256) with bf16 WMMA ----
        v8f acc[8];
        #pragma unroll
        for (int q = 0; q < 8; ++q) acc[q] = (v8f)0.0f;

        #pragma unroll 1
        for (int ks = 0; ks < 8; ++ks) {
            const int arow = mt * 16 + lrow;
            const int acol = ks * 32 + half * 8;
            v8bf alo = *(const v8bf*)&pbuf[arow * PSTR + acol];
            v8bf ahi = *(const v8bf*)&pbuf[arow * PSTR + acol + 16];
            v16bf afrag = join8(alo, ahi);
            // load all 8 B-fragments for this K-step (one coalesced clause) ...
            v16bf bf[8];
            #pragma unroll
            for (int nt2 = 0; nt2 < 8; ++nt2) {
                const int nt = nhalf * 8 + nt2;
                bf[nt2] = *(const v16bf*)&wswz[(((ks * 16 + nt) * 32) + lane) * 16];
            }
            // ... then 8 back-to-back WMMAs
            #pragma unroll
            for (int nt2 = 0; nt2 < 8; ++nt2) {
                acc[nt2] = __builtin_amdgcn_wmma_f32_16x16x32_bf16(
                    false, afrag, false, bf[nt2], (short)0, acc[nt2], false, false);
            }
        }
        __syncthreads();   // everyone done reading pbuf

        // ---- bias + tanh, store a back into pbuf as bf16 ----
        #pragma unroll
        for (int nt2 = 0; nt2 < 8; ++nt2) {
            const int o = nbase + nt2 * 16 + lrow;
            const float bias = att_proj_b[o];
            #pragma unroll
            for (int cv = 0; cv < 8; ++cv) {
                const int row = mt * 16 + half * 8 + cv;
                pbuf[row * PSTR + o] = f2bf(tanhf(acc[nt2][cv] + bias));
            }
        }
        __syncthreads();

        // ---- GEMM2: (64 x 256) @ (256 x 16[pad of 8]) -> logits, mask, store ----
        if (wave < 4) {
            const int mt2 = wave;
            v8f acc2 = (v8f)0.0f;
            #pragma unroll 1
            for (int ks = 0; ks < 8; ++ks) {
                const int arow = mt2 * 16 + lrow;
                const int acol = ks * 32 + half * 8;
                v8bf alo = *(const v8bf*)&pbuf[arow * PSTR + acol];
                v8bf ahi = *(const v8bf*)&pbuf[arow * PSTR + acol + 16];
                v16bf afrag = join8(alo, ahi);
                v16bf bfrag = *(const v16bf*)&awswz[((ks * 32) + lane) * 16];
                acc2 = __builtin_amdgcn_wmma_f32_16x16x32_bf16(
                    false, afrag, false, bfrag, (short)0, acc2, false, false);
            }
            const int h = lrow;
            if (h < Hc) {
                #pragma unroll
                for (int cv = 0; cv < 8; ++cv) {
                    const int k  = k0 + mt2 * 16 + half * 8 + cv;
                    const int hi = (j > k) ? j : k;
                    const int lo = (j > k) ? k : j;
                    const int rng = sp[hi + 1] - sp[lo];
                    const float mk = (rng == 0 || j == k) ? 1.0f : 0.0f;
                    attL[k * Hc + h] = acc2[cv] * mk;
                }
            }
        }
        __syncthreads();
    }

    // ---- softmax over k per head: wave w owns head w ----
    if (wave < Hc) {
        const int h = wave;
        float vv[8];
        float mx = -1e30f;
        #pragma unroll
        for (int i = 0; i < 8; ++i) { vv[i] = attL[(lane + i * 32) * Hc + h]; mx = fmaxf(mx, vv[i]); }
        for (int off = 16; off > 0; off >>= 1) mx = fmaxf(mx, __shfl_xor(mx, off, 32));
        float s = 0.0f;
        #pragma unroll
        for (int i = 0; i < 8; ++i) { vv[i] = __expf(vv[i] - mx); s += vv[i]; }
        for (int off = 16; off > 0; off >>= 1) s += __shfl_xor(s, off, 32);
        const float inv = 1.0f / s;
        #pragma unroll
        for (int i = 0; i < 8; ++i) attL[(lane + i * 32) * Hc + h] = vv[i] * inv;
    }
    __syncthreads();

    // ---- aggregation: x1[d][h] = sum_k attw[k][h] * x[b][k][d]; thread owns d=tid ----
    {
        float agg[Hc];
        #pragma unroll
        for (int h = 0; h < Hc; ++h) agg[h] = 0.0f;
        const float* xb = x + (b * Tc) * Dc;
        #pragma unroll 1
        for (int k = 0; k < Tc; ++k) {
            if (k + 8 < Tc) __builtin_prefetch(&xb[(k + 8) * Dc + tid], 0, 0);
            const float xv = xb[k * Dc + tid];
            #pragma unroll
            for (int h = 0; h < Hc; ++h) agg[h] += attL[k * Hc + h] * xv;
        }
        #pragma unroll
        for (int h = 0; h < Hc; ++h) x1s[tid * Hc + h] = agg[h];
    }
    __syncthreads();

    // ---- final projections: thread owns output column o=tid ----
    {
        const int o = tid;
        float accO = proj_att_b[o] + proj_no_b[o];
        #pragma unroll 1
        for (int i = 0; i < Dc * Hc; ++i)
            accO += x1s[i] * proj_att_w[i * Oc + o];
        #pragma unroll 1
        for (int d = 0; d < Dc; ++d)
            accO += xj[d] * proj_no_w[d * Oc + o];
        yout[(b * Tc + j) * Oc + o] = accO;
    }
}

// ============ BN stats ============
__global__ void k_zero_stats(float* __restrict__ sum, float* __restrict__ sumsq) {
    int t = threadIdx.x;
    if (t < Oc) { sum[t] = 0.0f; sumsq[t] = 0.0f; }
}

__global__ void k_stats(const float* __restrict__ y, float* __restrict__ sum, float* __restrict__ sumsq) {
    const int r0 = blockIdx.x * 4;
    const int o  = threadIdx.x;
    float s = 0.0f, s2 = 0.0f;
    #pragma unroll
    for (int rr = 0; rr < 4; ++rr) {
        float v = y[(r0 + rr) * Oc + o];
        s += v; s2 += v * v;
    }
    atomicAdd(&sum[o], s);
    atomicAdd(&sumsq[o], s2);
}

__global__ void k_bn_selu(const float* __restrict__ y,
                          const float* __restrict__ sum, const float* __restrict__ sumsq,
                          const float* __restrict__ gamma, const float* __restrict__ beta,
                          float* __restrict__ out) {
    const int row = blockIdx.x;
    const int o   = threadIdx.x;
    const float n   = (float)BT;
    const float mu  = sum[o] / n;
    const float var = sumsq[o] / n - mu * mu;
    float v = gamma[o] * (y[row * Oc + o] - mu) * rsqrtf(var + BN_EPS) + beta[o];
    v = (v > 0.0f) ? SELU_SCALE * v : SELU_SCALE * SELU_ALPHA * (__expf(v) - 1.0f);
    out[row * Oc + o] = v;
}

// ============ launch ============
extern "C" void kernel_launch(void* const* d_in, const int* in_sizes, int n_in,
                              void* d_out, int out_size, void* d_ws, size_t ws_size,
                              hipStream_t stream) {
    const float* x          = (const float*)d_in[0];
    const int*   boundary   = (const int*)d_in[1];
    const float* att_proj_w = (const float*)d_in[2];
    const float* att_proj_b = (const float*)d_in[3];
    const float* att_weight = (const float*)d_in[4];
    const float* proj_att_w = (const float*)d_in[5];
    const float* proj_att_b = (const float*)d_in[6];
    const float* proj_no_w  = (const float*)d_in[7];
    const float* proj_no_b  = (const float*)d_in[8];
    const float* bn_gamma   = (const float*)d_in[9];
    const float* bn_beta    = (const float*)d_in[10];

    char* ws = (char*)d_ws;
    int*    spad  = (int*)   (ws + WS_SPAD);
    __bf16* wswz  = (__bf16*)(ws + WS_WSWZ);
    __bf16* awswz = (__bf16*)(ws + WS_AWSWZ);
    float*  y     = (float*) (ws + WS_Y);
    float*  sum   = (float*) (ws + WS_SUM);
    float*  sumsq = (float*) (ws + WS_SUMSQ);

    k_prep_spad<<<Bc, 32, 0, stream>>>(boundary, spad);
    k_prep_w<<<(65536 + 4096) / 256, 256, 0, stream>>>(att_proj_w, att_weight, wswz, awswz);
    k_attn<<<BT, 256, 0, stream>>>(x, att_proj_b, proj_att_w, proj_att_b,
                                   proj_no_w, proj_no_b, wswz, awswz, spad, y);
    k_zero_stats<<<1, 256, 0, stream>>>(sum, sumsq);
    k_stats<<<BT / 4, 256, 0, stream>>>(y, sum, sumsq);
    k_bn_selu<<<BT, 256, 0, stream>>>(y, sum, sumsq, bn_gamma, bn_beta, (float*)d_out);
}
